// GAT_19688130085104
// MI455X (gfx1250) — matrix operand order repaired and checked by reading, not compiled, not run on hardware
//
#include <hip/hip_runtime.h>
#include <hip/hip_bf16.h>
#include <stdint.h>

// ---------------------------------------------------------------------------
// 3-layer GAT for MI455X (gfx1250, wave32).
// Dense node transforms: v_wmma_f32_16x16x32_bf16 (16x16 tile / wave).
// Edge softmax/scatter: fp32, b128 gathers, one atomicAdd per output dword,
// alpha normalized once per (edge,head) instead of per feature.
// ---------------------------------------------------------------------------

typedef __attribute__((ext_vector_type(16))) __bf16 v16bf;
typedef __attribute__((ext_vector_type(8)))  __bf16 v8bf;
typedef __attribute__((ext_vector_type(8)))  float  v8f;

#define NEG_SLOPE 0.2f

__device__ __forceinline__ float leaky(float x) { return x > 0.f ? x : NEG_SLOPE * x; }

// round-to-nearest-even f32 -> bf16 raw bits
__device__ __forceinline__ unsigned short f32_bf16(float f) {
  unsigned int u = __float_as_uint(f);
  u += 0x7fffu + ((u >> 16) & 1u);
  return (unsigned short)(u >> 16);
}

// order-preserving float <-> uint32 so segment-max can use integer atomicMax
__device__ __forceinline__ unsigned int encf(float f) {
  unsigned int u = __float_as_uint(f);
  return (u & 0x80000000u) ? ~u : (u | 0x80000000u);
}
__device__ __forceinline__ float decf(unsigned int u) {
  return __uint_as_float((u & 0x80000000u) ? (u ^ 0x80000000u) : ~u);
}

// --------------------------- conversion (b128 in / b64 out) ----------------
__global__ void k_f32_to_bf16_v4(const float* __restrict__ src,
                                 unsigned short* __restrict__ dst, int quads) {
  int i = blockIdx.x * blockDim.x + threadIdx.x;
  if (i >= quads) return;
  float4 v = reinterpret_cast<const float4*>(src)[i];
  ushort4 o;
  o.x = f32_bf16(v.x); o.y = f32_bf16(v.y);
  o.z = f32_bf16(v.z); o.w = f32_bf16(v.w);
  reinterpret_cast<ushort4*>(dst)[i] = o;
}

// --------------------------- WMMA GEMM -------------------------------------
// H[N,Fout] = Xb[N,K](bf16) @ Wb[K,Fout](bf16), fp32 accumulate.
// One wave per 16x16 tile; blockDim=256 (8 waves), wave w -> column tile w.
// A fragment (16-bit A 16x32, ISA 7.12.2): lane holds row M=lane&15,
//   elems 0..7 = K kb+0..7, elems 8..15 = K kb+16..23, kb = (lane>>4)*8.
// B fragment: lane holds column N=lane&15, elems 0..7 = K kb+0..7,
//   elems 8..15 = K kb+16..23 (same K order as A); strided scalar loads.
__global__ void k_gemm_wmma(const unsigned short* __restrict__ Xb,
                            const unsigned short* __restrict__ Wb,
                            float* __restrict__ H,
                            int K, int Fout, int colTiles) {
  const int lane = threadIdx.x & 31;
  const int wave = threadIdx.x >> 5;
  if (wave >= colTiles) return;

  const int rowBase = blockIdx.x * 16;
  const int r    = lane & 15;
  const int half = lane >> 4;      // 0 or 1
  const int kb   = half * 8;
  const int colBase = wave * 16;

  const __bf16* __restrict__ arow =
      reinterpret_cast<const __bf16*>(Xb) + (size_t)(rowBase + r) * K;
  const __bf16* __restrict__ wbase = reinterpret_cast<const __bf16*>(Wb);

  v8f acc = {};
  for (int k0 = 0; k0 < K; k0 += 32) {
    union { v16bf v; v8bf h[2]; } A, B;
    A.h[0] = *reinterpret_cast<const v8bf*>(arow + k0 + kb);
    A.h[1] = *reinterpret_cast<const v8bf*>(arow + k0 + kb + 16);
    const __bf16* bcol = wbase + (size_t)(k0 + kb) * Fout + colBase + r;
#pragma unroll
    for (int s = 0; s < 8; ++s) {
      B.v[s] = bcol[(size_t)s * Fout];
      B.v[8 + s] = bcol[(size_t)(16 + s) * Fout];
    }
    acc = __builtin_amdgcn_wmma_f32_16x16x32_bf16(
        false, A.v, false, B.v, (short)0, acc, false, false);
  }

  // C/D layout: VGPR i -> M = i + half*8, N = r
  float* out = H + (size_t)(rowBase + half * 8) * Fout + colBase + r;
#pragma unroll
  for (int i = 0; i < 8; ++i) out[(size_t)i * Fout] = acc[i];
}

// --------------------------- attention logits ------------------------------
__global__ void k_att(const float* __restrict__ H,
                      const float* __restrict__ attS,
                      const float* __restrict__ attD,
                      float* __restrict__ aS, float* __restrict__ aD,
                      int Nn, int heads, int ch, int F) {
  int i = blockIdx.x * blockDim.x + threadIdx.x;
  if (i >= Nn * heads) return;
  int n = i / heads, hd = i % heads;
  const float* hp = H + (size_t)n * F + hd * ch;
  const float* as = attS + hd * ch;
  const float* ad = attD + hd * ch;
  float s = 0.f, d = 0.f;
  for (int c = 0; c < ch; ++c) { float v = hp[c]; s += v * as[c]; d += v * ad[c]; }
  aS[i] = s; aD[i] = d;
}

// --------------------------- edge passes -----------------------------------
__device__ __forceinline__ void edge_ends(const int* __restrict__ ei, int e, int E,
                                          int& s, int& d) {
  if (e < E) { s = ei[e]; d = ei[E + e]; } else { s = e - E; d = s; }
}

// (A) leaky logit + segment max (order-preserving uint atomicMax)
__global__ void k_edgeA(const int* __restrict__ ei,
                        const float* __restrict__ aS, const float* __restrict__ aD,
                        float* __restrict__ ecoef, unsigned int* __restrict__ mEnc,
                        int E, int Nn, int heads) {
  int i = blockIdx.x * blockDim.x + threadIdx.x;
  if (i >= (E + Nn) * heads) return;
  int e = i / heads, hd = i % heads, s, d;
  edge_ends(ei, e, E, s, d);
  float v = leaky(aS[s * heads + hd] + aD[d * heads + hd]);
  ecoef[i] = v;
  atomicMax(&mEnc[d * heads + hd], encf(v));
}

// (B) exp(e - m) + segment sum
__global__ void k_edgeB(const int* __restrict__ ei,
                        float* __restrict__ ecoef,
                        const unsigned int* __restrict__ mEnc,
                        float* __restrict__ ssum,
                        int E, int Nn, int heads) {
  int i = blockIdx.x * blockDim.x + threadIdx.x;
  if (i >= (E + Nn) * heads) return;
  int e = i / heads, hd = i % heads, s, d;
  edge_ends(ei, e, E, s, d);
  float m = decf(mEnc[d * heads + hd]);
  float w = __expf(ecoef[i] - m);
  ecoef[i] = w;
  atomicAdd(&ssum[d * heads + hd], w);
}

// (B2) normalize once per (edge,head): ecoef <- alpha
__global__ void k_alpha(const int* __restrict__ ei,
                        float* __restrict__ ecoef,
                        const float* __restrict__ ssum,
                        int E, int Nn, int heads) {
  int i = blockIdx.x * blockDim.x + threadIdx.x;
  if (i >= (E + Nn) * heads) return;
  int e = i / heads, hd = i % heads, s, d;
  edge_ends(ei, e, E, s, d);
  ecoef[i] = ecoef[i] / (ssum[d * heads + hd] + 1e-16f);
}

// (C) alpha-weighted message scatter: one thread per (edge, 4-feature quad).
// b128 gather of h[src], 4 fp32 atomic adds into agg[dst].
__global__ void k_edgeC(const int* __restrict__ ei,
                        const float* __restrict__ H,
                        const float* __restrict__ alpha,
                        float* __restrict__ agg,
                        int E, int Nn, int heads, int ch, int F) {
  const int quads = F >> 2;
  long long i = (long long)blockIdx.x * blockDim.x + threadIdx.x;
  long long total = (long long)(E + Nn) * quads;
  if (i >= total) return;
  int e = (int)(i / quads);
  int f = ((int)(i % quads)) << 2;      // quad never straddles a head (ch%4==0)
  int hd = f / ch, s, d;
  edge_ends(ei, e, E, s, d);
  const float* hp = H + (size_t)s * F + f;
  __builtin_prefetch(hp, 0, 0);         // global_prefetch_b8 on the gather line
  float a = alpha[(size_t)e * heads + hd];
  float4 hv = *reinterpret_cast<const float4*>(hp);
  float* out = agg + (size_t)d * F + f;
  atomicAdd(out + 0, hv.x * a);
  atomicAdd(out + 1, hv.y * a);
  atomicAdd(out + 2, hv.z * a);
  atomicAdd(out + 3, hv.w * a);
}

// --------------------------- bias + leaky (b128) ---------------------------
__global__ void k_final_v4(float* __restrict__ agg, const float* __restrict__ b,
                           float* __restrict__ out, int Nn, int F) {
  int quads = (Nn * F) >> 2;
  int i = blockIdx.x * blockDim.x + threadIdx.x;
  if (i >= quads) return;
  int f = (i << 2) % F;                 // 16B-aligned bias offset
  float4 v = reinterpret_cast<const float4*>(agg)[i];
  float4 bb = *reinterpret_cast<const float4*>(b + f);
  v.x = leaky(v.x + bb.x);
  v.y = leaky(v.y + bb.y);
  v.z = leaky(v.z + bb.z);
  v.w = leaky(v.w + bb.w);
  if (out) reinterpret_cast<float4*>(out)[i] = v;
  else     reinterpret_cast<float4*>(agg)[i] = v;
}

// ---------------------------------------------------------------------------
extern "C" void kernel_launch(void* const* d_in, const int* in_sizes, int n_in,
                              void* d_out, int out_size, void* d_ws, size_t ws_size,
                              hipStream_t stream) {
  const float* x   = (const float*)d_in[0];
  const int*   ei  = (const int*)  d_in[1];
  const float* W0  = (const float*)d_in[2];
  const float* as0 = (const float*)d_in[3];
  const float* ad0 = (const float*)d_in[4];
  const float* b0  = (const float*)d_in[5];
  const float* W1  = (const float*)d_in[6];
  const float* as1 = (const float*)d_in[7];
  const float* ad1 = (const float*)d_in[8];
  const float* b1  = (const float*)d_in[9];
  const float* W2  = (const float*)d_in[10];
  const float* as2 = (const float*)d_in[11];
  const float* ad2 = (const float*)d_in[12];
  const float* b2  = (const float*)d_in[13];

  const int N    = in_sizes[0] / 128;   // 50000 (multiple of 16)
  const int E    = in_sizes[1] / 2;     // 800000
  const int Etot = E + N;

  // workspace carve (≈81 MB total)
  char* w = (char*)d_ws;
  auto carve = [&](size_t bytes) {
    void* p = (void*)w;
    w += (bytes + 255) & ~(size_t)255;
    return p;
  };
  unsigned short* xb   = (unsigned short*)carve((size_t)N * 128 * 2);
  unsigned short* wbf  = (unsigned short*)carve((size_t)128 * 128 * 2);
  float*          h    = (float*)carve((size_t)N * 128 * 4);
  float*          agg  = (float*)carve((size_t)N * 128 * 4);
  float*          aS   = (float*)carve((size_t)N * 4 * 4);
  float*          aD   = (float*)carve((size_t)N * 4 * 4);
  unsigned int*   mEnc = (unsigned int*)carve((size_t)N * 4 * 4);
  float*          ssum = (float*)carve((size_t)N * 4 * 4);
  float*          ecoef= (float*)carve((size_t)Etot * 4 * 4);

  auto runLayer = [&](const float* Xin, const float* Wl, const float* asl,
                      const float* adl, const float* bl,
                      int K, int F, int heads, float* outPtr) {
    const int ch = F / heads;
    // bf16 conversion (X first: layers 1/2 read Xin aliased with agg)
    {
      int q = (N * K) >> 2;
      k_f32_to_bf16_v4<<<(q + 255) / 256, 256, 0, stream>>>(Xin, xb, q);
      int qw = (K * F) >> 2;
      k_f32_to_bf16_v4<<<(qw + 255) / 256, 256, 0, stream>>>(Wl, wbf, qw);
    }
    // reset accumulators (after conversion, stream-ordered)
    hipMemsetAsync(agg,  0, (size_t)N * F * 4,     stream);
    hipMemsetAsync(mEnc, 0, (size_t)N * heads * 4, stream);  // 0 == encoded -inf
    hipMemsetAsync(ssum, 0, (size_t)N * heads * 4, stream);
    // h = Xb @ Wb  (WMMA)
    k_gemm_wmma<<<N / 16, 256, 0, stream>>>(xb, wbf, h, K, F, F / 16);
    // per-(node,head) logits
    {
      int t = N * heads;
      k_att<<<(t + 255) / 256, 256, 0, stream>>>(h, asl, adl, aS, aD, N, heads, ch, F);
    }
    // segment softmax + message scatter
    {
      int t = Etot * heads;
      k_edgeA<<<(t + 255) / 256, 256, 0, stream>>>(ei, aS, aD, ecoef, mEnc, E, N, heads);
      k_edgeB<<<(t + 255) / 256, 256, 0, stream>>>(ei, ecoef, mEnc, ssum, E, N, heads);
      k_alpha<<<(t + 255) / 256, 256, 0, stream>>>(ei, ecoef, ssum, E, N, heads);
      long long tc = (long long)Etot * (F >> 2);
      k_edgeC<<<(unsigned)((tc + 255) / 256), 256, 0, stream>>>(
          ei, h, ecoef, agg, E, N, heads, ch, F);
    }
    // bias + leaky (in-place -> agg becomes next layer's input, or to d_out)
    {
      int q = (N * F) >> 2;
      k_final_v4<<<(q + 255) / 256, 256, 0, stream>>>(agg, bl, outPtr, N, F);
    }
  };

  runLayer(x,   W0, as0, ad0, b0, 128, 128, 4, nullptr);
  runLayer(agg, W1, as1, ad1, b1, 128, 128, 4, nullptr);
  runLayer(agg, W2, as2, ad2, b2, 128,  64, 1, (float*)d_out);
}
